// nonlocalblock_38689065402520
// MI455X (gfx1250) — matrix operand order repaired and checked by reading, not compiled, not run on hardware
//
#include <hip/hip_runtime.h>
#include <hip/hip_bf16.h>

typedef __attribute__((ext_vector_type(16))) _Float16 v16h;
typedef __attribute__((ext_vector_type(8)))  float    v8f;

#define BATCH 8
#define CCH   64
#define HW    512
#define KPOOL 8
#define NP    4096        // pooled N = 64*64
#define LOG2E 1.44269504088896340736f

static __device__ __forceinline__ v8f wmma_f16(v16h a, v16h b, v8f c) {
    // D = A(16x32 f16) * B(32x16 f16) + C(16x16 f32)
    return __builtin_amdgcn_wmma_f32_16x16x32_f16(
        false, a, false, b, (short)0, c, false, false);
}

static __device__ __forceinline__ float exp2v(float x) {
    return __builtin_amdgcn_exp2f(x);             // raw v_exp_f32
}

// single-instruction max (avoids maxnum canonicalization self-max pairs)
static __device__ __forceinline__ float vmax(float a, float b) {
    float d;
    asm("v_max_num_f32 %0, %1, %2" : "=v"(d) : "v"(a), "v"(b));
    return d;
}

// butterfly max over each 16-lane DPP row: 4 x v_max_num_f32_dpp, pure VALU
static __device__ __forceinline__ float rowmax16(float m) {
    float t;
    asm("v_max_num_f32_dpp %0, %1, %1 quad_perm:[1,0,3,2] row_mask:0xf bank_mask:0xf bound_ctrl:1"
        : "=v"(t) : "v"(m));
    asm("v_max_num_f32_dpp %0, %1, %1 quad_perm:[2,3,0,1] row_mask:0xf bank_mask:0xf bound_ctrl:1"
        : "=v"(m) : "v"(t));
    asm("v_max_num_f32_dpp %0, %1, %1 row_half_mirror row_mask:0xf bank_mask:0xf bound_ctrl:1"
        : "=v"(t) : "v"(m));
    asm("v_max_num_f32_dpp %0, %1, %1 row_mirror row_mask:0xf bank_mask:0xf bound_ctrl:1"
        : "=v"(m) : "v"(t));
    return m;
}

// ---------------------------------------------------------------------------
// Kernel 1: 8x8 average pool  x[B,C,512,512] -> u[B,C,4096]
// ---------------------------------------------------------------------------
__global__ void pool_kernel(const float* __restrict__ x, float* __restrict__ u) {
    int idx = blockIdx.x * 256 + threadIdx.x;      // B*C*64*64 = 2,097,152
    int wo = idx & 63;
    int ho = (idx >> 6) & 63;
    int bc = idx >> 12;                            // b*64 + c
    const float* src = x + ((size_t)bc * HW + (size_t)ho * KPOOL) * HW + wo * KPOOL;
    float s = 0.f;
#pragma unroll
    for (int r = 0; r < 8; ++r) {
        const float4* p = (const float4*)(src + (size_t)r * HW);
        float4 a = p[0], b4 = p[1];
        s += a.x + a.y + a.z + a.w + b4.x + b4.y + b4.z + b4.w;
    }
    u[idx] = s * (1.0f / 64.0f);
}

// ---------------------------------------------------------------------------
// Kernel 2: 1x1 conv projections -> WMMA fragment layouts (f16)
//   thetaA: A-frags, pre-scaled by log2(e)   [b][ntile(256)][lane(32)][32 halves]
//   phiB  : B-frags (K = channel)            [b][mtile(256)][lane(32)][32 halves]
//   gB    : B-frags (K = position), K-permuted by pi^-1 to match the packed
//           P transpose in the attention kernel  [b][mchunk(128)][ct(4)][lane][16]
// ---------------------------------------------------------------------------
__global__ void proj_kernel(const float* __restrict__ u,
                            const float* __restrict__ Wt, const float* __restrict__ bt,
                            const float* __restrict__ Wp, const float* __restrict__ bp,
                            const float* __restrict__ Wg, const float* __restrict__ bg,
                            _Float16* __restrict__ thetaA,
                            _Float16* __restrict__ phiB,
                            _Float16* __restrict__ gB) {
    int idx  = blockIdx.x * 256 + threadIdx.x;     // ((b*3+proj)*64+o)*4096+n
    int n    = idx & 4095;
    int o    = (idx >> 12) & 63;
    int bp3  = idx >> 18;
    int proj = bp3 % 3;
    int b    = bp3 / 3;

    const float* W; const float* bias;
    if      (proj == 0) { W = Wt; bias = bt; }
    else if (proj == 1) { W = Wp; bias = bp; }
    else                { W = Wg; bias = bg; }

    const float* uc = u + (size_t)b * CCH * NP + n;
    float s = bias[o];
#pragma unroll 8
    for (int c = 0; c < CCH; ++c)
        s += W[o * CCH + c] * uc[(size_t)c * NP];

    if (proj == 0) {
        // A-layout: lane = hi*16 + (n%16); half = c0*16 + p*2 + e
        _Float16 h = (_Float16)(s * LOG2E);        // base-2 softmax domain
        int t  = n >> 4, m = n & 15;
        int c0 = o >> 5, kk = o & 31;
        int p  = ((kk >> 4) << 2) | ((kk >> 1) & 3);
        int hi = (kk >> 3) & 1;
        int lane = (hi << 4) | m;
        int half = (c0 << 4) + (p << 1) + (kk & 1);
        thetaA[(((size_t)b * 256 + t) * 32 + lane) * 32 + half] = h;
    } else if (proj == 1) {
        // B-layout (N = position m = n, K = channel o)
        _Float16 h = (_Float16)s;
        int mt = n >> 4;
        int c0 = o >> 5, kk = o & 31;
        int hi = (kk >> 3) & 1, v = ((kk >> 4) << 2) | ((kk >> 1) & 3), e = kk & 1;
        int lane = (hi << 4) | (n & 15);
        int half = (c0 << 4) + (v << 1) + e;
        phiB[(((size_t)b * 256 + mt) * 32 + lane) * 32 + half] = h;
    } else {
        // B-layout (N = channel o, K = position n), with K remapped through
        // pi^-1 so it matches P stored as packed (col j, col j+16) dwords:
        //   col(K) = (K&1)*16 + ((K>>4)<<3) + ((K>>1)&7)
        _Float16 h = (_Float16)s;
        int mc = n >> 5, km = n & 31;
        int e2 = (km >> 4) & 1, h2 = (km >> 3) & 1, p2 = km & 7;
        int kw = (h2 << 4) | ((p2 >> 2) << 3) | ((p2 & 3) << 1) | e2;  // pi^-1(km)
        int hi = kw >> 4, v = (kw >> 1) & 7, e = kw & 1;
        int lane = (hi << 4) | (o & 15);
        int ct = o >> 4;
        int half = (v << 1) + e;
        gB[((((size_t)b * 128 + mc) * 4 + ct) * 32 + lane) * 16 + half] = h;
    }
}

// ---------------------------------------------------------------------------
// Kernel 3: flash attention. 1 wave per 16-row tile, 8 waves/block.
//   S' = theta'.phi (base-2 domain); online softmax with DPP row-max,
//   rowsum via P.ones WMMA; y = P.g with K-permuted g fragments.
// ---------------------------------------------------------------------------
__global__ void __launch_bounds__(256)
attn_kernel(const v16h* __restrict__ thetaA,
            const v16h* __restrict__ phiB,
            const v16h* __restrict__ gB,
            float* __restrict__ y) {
    __shared__ __align__(16) unsigned lds32[8 * 256];     // 1KB per wave
    const int lane  = threadIdx.x & 31;
    const int wave  = threadIdx.x >> 5;
    const int tileG = blockIdx.x * 8 + wave;              // b*256 + t
    const int b     = tileG >> 8;
    const int t     = tileG & 255;
    unsigned* pbuf  = lds32 + wave * 256;                 // [row(16)][pair(16)] dwords

    // theta A-fragments for this 16-row tile (K = 0..31, 32..63)
    const v16h* tA = thetaA + ((size_t)tileG * 32 + lane) * 2;
    v16h A0 = tA[0];
    v16h A1 = tA[1];

    const _Float16 one1 = (_Float16)1.0f;
    const v16h ONES = { one1, one1, one1, one1, one1, one1, one1, one1,
                        one1, one1, one1, one1, one1, one1, one1, one1 };

    float runmax[8];
    v8f acc[4] = {};
    v8f accS = {};                                        // row sums, C/D layout
#pragma unroll
    for (int r = 0; r < 8; ++r) runmax[r] = -__builtin_inff();

    const int m16 = lane & 15;
    const int hi  = (lane >> 4) & 1;
    const int rowb = hi << 3;

    for (int mc = 0; mc < 128; ++mc) {
        // ---- S = theta . phi for 32 columns (two 16-col tiles) ----
        const v16h* pB0 = phiB + ((size_t)(b * 256 + mc * 2)     * 32 + lane) * 2;
        const v16h* pB1 = phiB + ((size_t)(b * 256 + mc * 2 + 1) * 32 + lane) * 2;
        v8f S0 = {}, S1 = {};
        S0 = wmma_f16(A0, pB0[0], S0);
        S0 = wmma_f16(A1, pB0[1], S0);
        S1 = wmma_f16(A0, pB1[0], S1);
        S1 = wmma_f16(A1, pB1[1], S1);

        // ---- row max via DPP butterflies ----
        float nm[8];
        int changed = 0;
#pragma unroll
        for (int r = 0; r < 8; ++r) {
            float m2 = vmax(S0[r], S1[r]);
            m2 = rowmax16(m2);
            changed |= (m2 > runmax[r]);
            nm[r] = vmax(runmax[r], m2);
        }
        // ---- rescale accumulators only when a row max advanced (rare) ----
        if (__ballot(changed)) {
#pragma unroll
            for (int r = 0; r < 8; ++r) {
                float sc = exp2v(runmax[r] - nm[r]);
                accS[r] *= sc;
#pragma unroll
                for (int ct = 0; ct < 4; ++ct)
                    acc[ct][r] *= sc;
            }
        }
        // ---- P = 2^(S - max), packed transpose into LDS ----
#pragma unroll
        for (int r = 0; r < 8; ++r) {
            runmax[r] = nm[r];
            float p0 = exp2v(S0[r] - nm[r]);
            float p1 = exp2v(S1[r] - nm[r]);
            auto pk = __builtin_amdgcn_cvt_pkrtz(p0, p1); // (col m16, col m16+16)
            pbuf[(rowb + r) * 16 + m16] = __builtin_bit_cast(unsigned, pk);
        }
        // A-fragment load: 8 contiguous dwords; K arrives pi-permuted, which
        // the g fragments (and the all-ones B) already match.
        v16h PA = *(const v16h*)(pbuf + m16 * 16 + hi * 8);

        // ---- rowsum += P . ones ; y += P . g ----
        accS = wmma_f16(PA, ONES, accS);
        const v16h* gb = gB + (((size_t)(b * 128 + mc)) * 4) * 32 + lane;
#pragma unroll
        for (int ct = 0; ct < 4; ++ct)
            acc[ct] = wmma_f16(PA, gb[(size_t)ct * 32], acc[ct]);
    }

    // ---- epilogue: normalize and store y[b][n][c] (f32) ----
    float inv[8];
#pragma unroll
    for (int r = 0; r < 8; ++r) inv[r] = 1.0f / accS[r];
#pragma unroll
    for (int ct = 0; ct < 4; ++ct) {
#pragma unroll
        for (int r = 0; r < 8; ++r) {
            int n = t * 16 + rowb + r;
            int c = ct * 16 + m16;
            y[((size_t)b * NP + n) * CCH + c] = acc[ct][r] * inv[r];
        }
    }
}

// ---------------------------------------------------------------------------
// Kernel 4: out-conv + nearest x8 upsample
// ---------------------------------------------------------------------------
__global__ void out_kernel(const float* __restrict__ y,
                           const float* __restrict__ Wc, const float* __restrict__ bc,
                           float* __restrict__ out) {
    int idx = blockIdx.x * 256 + threadIdx.x;      // B*C*4096 = 2,097,152
    int n = idx & 4095;
    int o = (idx >> 12) & 63;
    int b = idx >> 18;
    const float4* yr = (const float4*)(y + ((size_t)b * NP + n) * CCH);
    float s = bc[o];
#pragma unroll
    for (int c4 = 0; c4 < 16; ++c4) {
        float4 v  = yr[c4];
        const float* w = Wc + o * CCH + c4 * 4;
        s += w[0] * v.x + w[1] * v.y + w[2] * v.z + w[3] * v.w;
    }
    int hh = n >> 6, ww = n & 63;
    float4 v4 = make_float4(s, s, s, s);
    float* dst = out + ((size_t)(b * CCH + o) * HW + (size_t)hh * KPOOL) * HW + ww * KPOOL;
#pragma unroll
    for (int r = 0; r < 8; ++r) {
        float4* p = (float4*)(dst + (size_t)r * HW);
        p[0] = v4; p[1] = v4;
    }
}

// ---------------------------------------------------------------------------
extern "C" void kernel_launch(void* const* d_in, const int* in_sizes, int n_in,
                              void* d_out, int out_size, void* d_ws, size_t ws_size,
                              hipStream_t stream) {
    const float* x  = (const float*)d_in[0];
    const float* Wt = (const float*)d_in[1];
    const float* bt = (const float*)d_in[2];
    const float* Wp = (const float*)d_in[3];
    const float* bp = (const float*)d_in[4];
    const float* Wg = (const float*)d_in[5];
    const float* bg = (const float*)d_in[6];
    const float* Wc = (const float*)d_in[7];
    const float* bc = (const float*)d_in[8];
    float* out = (float*)d_out;

    char* ws = (char*)d_ws;
    float*    u      = (float*)(ws);                          //  8,388,608 B
    _Float16* thetaA = (_Float16*)(ws + 8388608);             //  4,194,304 B
    _Float16* phiB   = (_Float16*)(ws + 12582912);            //  4,194,304 B
    _Float16* gB     = (_Float16*)(ws + 16777216);            //  4,194,304 B
    float*    yb     = (float*)(ws + 20971520);               //  8,388,608 B

    pool_kernel<<<8192, 256, 0, stream>>>(x, u);
    proj_kernel<<<24576, 256, 0, stream>>>(u, Wt, bt, Wp, bp, Wg, bg,
                                           thetaA, phiB, gB);
    attn_kernel<<<256, 256, 0, stream>>>((const v16h*)thetaA, (const v16h*)phiB,
                                         (const v16h*)gB, yb);
    out_kernel<<<8192, 256, 0, stream>>>(yb, Wc, bc, out);
}
